// BotUpSaliency_90340342104690
// MI455X (gfx1250) — matrix operand order repaired and checked by reading, not compiled, not run on hardware
//
#include <hip/hip_runtime.h>
#include <hip/hip_bf16.h>

typedef __attribute__((ext_vector_type(16))) _Float16 v16h;
typedef __attribute__((ext_vector_type(8)))  float    v8f;
typedef __attribute__((ext_vector_type(4)))  unsigned int v4u;

#define HH 512
#define WW 512
#define KC 12          // real channels
#define KP 16          // padded channels
#define BATCH 2
#define NPIX ((size_t)BATCH * HH * WW)
#define KS 15
#define TILE_W 16      // output pixels per wave-row (M)
#define TILE_H 16      // output rows per block (each wave owns rows wv and wv+8)
#define TW 31          // LDS tile width  (16 + 14 halo + 1 for tap-pair pad)
#define TR 30          // LDS tile height (16 + 14 halo)
#define WPK_ELEMS (15 * 8 * 32 * 16)   // per conv: rows * tap-pairs * lanes * halves

__device__ __forceinline__ int reflect(int i, int n) {
    // numpy 'symmetric' padding: -1 -> 0, -2 -> 1, n -> n-1, ...
    if (i < 0) return -1 - i;
    if (i >= n) return 2 * n - 1 - i;
    return i;
}

#define WMMA_F16(A, B, C) \
    __builtin_amdgcn_wmma_f32_16x16x32_f16(false, (A), false, (B), (short)0, (C), false, false)

// ---------------------------------------------------------------------------
// Pack W / J (15,15,12,12 f32, HWIO) and psi (12,12) into the exact CDNA5
// WMMA B-matrix lane layout for v_wmma_f32_16x16x32_f16:
//   lane L: g = L/16, n = L%16 is out-channel; half h<8 is channel 8g+h of
//   tap 2p, half h>=8 is channel 8g+h-8 of tap 2p+1 (channels >= 12 and
//   tap 15 padded with zero).
// ---------------------------------------------------------------------------
__global__ __launch_bounds__(256) void prep_kernel(
    const float* __restrict__ Wsrc, const float* __restrict__ Jsrc,
    const float* __restrict__ Psrc,
    _Float16* __restrict__ Wpk, _Float16* __restrict__ Jpk,
    _Float16* __restrict__ Ppk)
{
    int i = blockIdx.x * 256 + threadIdx.x;
    if (i < 2 * WPK_ELEMS) {
        const float* src = (i < WPK_ELEMS) ? Wsrc : Jsrc;
        _Float16*    dst = (i < WPK_ELEMS) ? Wpk  : Jpk;
        int j    = (i < WPK_ELEMS) ? i : i - WPK_ELEMS;
        int h    = j & 15;           // K row within lane (channel of the tap)
        int lane = (j >> 4) & 31;
        int rp   = j >> 9;           // r*8 + p
        int r    = rp >> 3;
        int p    = rp & 7;
        int g    = lane >> 4;
        int n    = lane & 15;        // output channel
        int tap  = 2 * p + (h >> 3); // kernel column
        int ch   = 8 * g + (h & 7);
        float v  = 0.0f;
        if (tap < KS && ch < KC && n < KC)
            v = src[((r * KS + tap) * KC + ch) * KC + n];
        dst[j] = (_Float16)v;
    } else {
        int j = i - 2 * WPK_ELEMS;
        if (j < 512) {               // 32 lanes * 16 halves
            int h = j & 15, lane = j >> 4;
            int g = lane >> 4, n = lane & 15;
            float v = 0.0f;
            if (h < 8 && 8 * g + h < KC && n < KC) v = Psrc[(8 * g + h) * KC + n];
            Ppk[j] = (_Float16)v;
        }
    }
}

// ---------------------------------------------------------------------------
// State init: x = 0.01, y = 1.0, out_acc = 0, both gx buffers = 0
// (gx0 = clip(0.01 - 1, 0, 1) = 0; padded channels stay 0 forever)
// ---------------------------------------------------------------------------
__global__ __launch_bounds__(256) void init_kernel(
    float* __restrict__ x, float* __restrict__ y, float* __restrict__ outAcc,
    _Float16* __restrict__ gxA, _Float16* __restrict__ gxB)
{
    size_t p = (size_t)blockIdx.x * 256 + threadIdx.x;
    if (p >= NPIX) return;
    for (int c = 0; c < KC; c++) {
        x[p * KC + c] = 0.01f;
        y[p * KC + c] = 1.0f;
        outAcc[p * KC + c] = 0.0f;
    }
    for (int c = 0; c < KP; c++) {
        gxA[p * KP + c] = (_Float16)0.0f;
        gxB[p * KP + c] = (_Float16)0.0f;
    }
}

// ---------------------------------------------------------------------------
// One recurrence step. Block = 256 threads = 8 waves; block tile = 16 rows x
// 16 cols of output pixels. Wave wv computes rows wv and wv+8 so each
// B-fragment (weights) is reused by two WMMAs -> half the L2 weight traffic.
// ---------------------------------------------------------------------------
__global__ __launch_bounds__(256) void step_kernel(
    const _Float16* __restrict__ gxIn,   // [B][H][W][16] f16, symmetric halo via reflect
    _Float16*       __restrict__ gxOut,
    float* __restrict__ xs, float* __restrict__ ys,
    float* __restrict__ outAcc,
    const float* __restrict__ inp,       // [B][H][W][12]
    const _Float16* __restrict__ Wpk,
    const _Float16* __restrict__ Jpk,
    const _Float16* __restrict__ Ppk)
{
    __shared__ __align__(16) _Float16 gxT[TR * TW * KP];          // 29.8 KB
    __shared__ __align__(16) _Float16 gyT[TILE_H * TILE_W * KP];  // 8 KB
    __shared__ float sT[20 * 20];        // s with +-2 halo (zero-pad at image edge)
    __shared__ float boxT[TILE_H * TILE_W];

    const int tid = threadIdx.x;
    const int x0 = blockIdx.x * TILE_W;
    const int y0 = blockIdx.y * TILE_H;
    const int b  = blockIdx.z;
    const size_t imgBase = (size_t)b * HH * WW;

    // --- stage gx halo tile via CDNA5 async global->LDS DMA (ASYNCcnt) ---
    for (int i = tid; i < TR * TW; i += 256) {
        int ty = i / TW, tx = i % TW;
        int gyi = reflect(y0 - 7 + ty, HH);
        int gxi = reflect(x0 - 7 + tx, WW);
        const _Float16* src = gxIn + (imgBase + (size_t)gyi * WW + gxi) * KP;
        _Float16* dst = gxT + (size_t)i * KP;
        asm volatile("global_load_async_to_lds_b128 %0, %1, off"
                     :: "v"((unsigned)(unsigned long long)dst), "v"(src)
                     : "memory");
        asm volatile("global_load_async_to_lds_b128 %0, %1, off"
                     :: "v"((unsigned)(unsigned long long)(dst + 8)), "v"(src + 8)
                     : "memory");
    }
    // --- stage gy = g(y) for this block's own pixels (feeds the psi WMMA) ---
    for (int i = tid; i < TILE_H * TILE_W * 4; i += 256) {       // zero pad chans 12..15
        int p = i >> 2, c = KC + (i & 3);
        gyT[p * KP + c] = (_Float16)0.0f;
    }
    for (int i = tid; i < TILE_H * TILE_W * KC; i += 256) {
        int p = i / KC, c = i % KC;
        int py = y0 + p / TILE_W, px = x0 + p % TILE_W;
        float yv = ys[(imgBase + (size_t)py * WW + px) * KC + c];
        float yc = fmaxf(yv, 0.0f);
        float gyv = (yc <= 1.2f) ? 0.21f * yc : 0.21f * 1.2f + 2.5f * (yc - 1.2f);
        gyT[p * KP + c] = (_Float16)gyv;
    }
    asm volatile("s_wait_asynccnt 0x0" ::: "memory");
    __syncthreads();

    // --- s = sum_c gx, with ZERO padding at image border (box conv is SAME) ---
    for (int i = tid; i < 20 * 20; i += 256) {
        int r = i / 20, c = i % 20;
        int gy_ = y0 + r - 2, gx_ = x0 + c - 2;
        float s = 0.0f;
        if (gy_ >= 0 && gy_ < HH && gx_ >= 0 && gx_ < WW) {
            const _Float16* p = gxT + ((r + 5) * TW + (c + 5)) * KP;
            #pragma unroll
            for (int k = 0; k < KP; k++) s += (float)p[k];
        }
        sT[i] = s;
    }
    __syncthreads();
    for (int i = tid; i < TILE_H * TILE_W; i += 256) {
        int r = i / TILE_W, c = i % TILE_W;
        float bs = 0.0f;
        #pragma unroll
        for (int a = 0; a < 5; a++)
            #pragma unroll
            for (int e = 0; e < 5; e++) bs += sT[(r + a) * 20 + (c + e)];
        boxT[i] = bs;
    }
    __syncthreads();

    // --- implicit-GEMM convolutions: M=16 pixels, N=16 (12 used), K=2 taps x 16ch ---
    const int wv   = tid >> 5;          // wave id; owns output rows wv and wv+8
    const int lane = tid & 31;
    const int g    = lane >> 4;         // K-half select
    const int m    = lane & 15;         // A-matrix row = pixel index

    v8f accW[2] = {};
    v8f accJ[2] = {};
    for (int r = 0; r < KS; r++) {
        const _Float16* row0 = gxT + (wv + r) * TW * KP;
        const _Float16* row1 = gxT + (wv + 8 + r) * TW * KP;
        #pragma unroll
        for (int p = 0; p < 8; p++) {
            const int dx0 = 2 * p, dx1 = 2 * p + 1;   // tap 15 = zero weights
            union U { v16h h; v4u u[2]; };
            U A0, A1, Bw, Bj;
            // A layout: lane g holds chans 8g..8g+7 of tap0 (K 0..7 / 8..15)
            //           and chans 8g..8g+7 of tap1 (K 16..23 / 24..31)
            A0.u[0] = *(const v4u*)(row0 + (m + dx0) * KP + 8 * g);
            A0.u[1] = *(const v4u*)(row0 + (m + dx1) * KP + 8 * g);
            A1.u[0] = *(const v4u*)(row1 + (m + dx0) * KP + 8 * g);
            A1.u[1] = *(const v4u*)(row1 + (m + dx1) * KP + 8 * g);
            const v4u* wb = (const v4u*)(Wpk + ((((r * 8 + p) * 32) + lane) << 4));
            Bw.u[0] = wb[0]; Bw.u[1] = wb[1];
            const v4u* jb = (const v4u*)(Jpk + ((((r * 8 + p) * 32) + lane) << 4));
            Bj.u[0] = jb[0]; Bj.u[1] = jb[1];
            accW[0] = WMMA_F16(A0.h, Bw.h, accW[0]);
            accW[1] = WMMA_F16(A1.h, Bw.h, accW[1]);
            accJ[0] = WMMA_F16(A0.h, Bj.h, accJ[0]);
            accJ[1] = WMMA_F16(A1.h, Bj.h, accJ[1]);
        }
    }

    // --- psi matmul: inhibs_psi[p, o] = sum_i gy[p, i] * psi[i, o] ---
    v8f accP[2] = {};
    {
        union U { v16h h; v4u u[2]; };
        U Bp;
        const v4u* pb = (const v4u*)(Ppk + (lane << 4));
        Bp.u[0] = pb[0]; Bp.u[1] = pb[1];
        v4u zero4 = {0u, 0u, 0u, 0u};
        #pragma unroll
        for (int rs = 0; rs < 2; rs++) {
            U Ag;
            Ag.u[0] = *(const v4u*)(gyT + ((wv + 8 * rs) * TILE_W + m) * KP + 8 * g);
            Ag.u[1] = zero4;                          // K 16..31 are zero
            accP[rs] = WMMA_F16(Ag.h, Bp.h, accP[rs]);
        }
    }

    // --- elementwise update. D layout: lane n = L%16 = channel; VGPR v = pixel 8g+v ---
    const int n  = lane & 15;
    const int mb = g * 8;
    if (n < KC) {
        #pragma unroll
        for (int rs = 0; rs < 2; rs++) {
            const int py = y0 + wv + 8 * rs;
            #pragma unroll
            for (int v = 0; v < 8; v++) {
                const int pm = mb + v;
                const int px = x0 + pm;
                const size_t pidx = imgBase + (size_t)py * WW + px;
                const size_t ci = pidx * KC + n;
                float xo = xs[ci], yo = ys[ci], in_v = inp[ci];
                float gxo = fminf(fmaxf(xo - 1.0f, 0.0f), 1.0f);
                float yc  = fmaxf(yo, 0.0f);
                float gyo = (yc <= 1.2f) ? 0.21f * yc
                                         : 0.21f * 1.2f + 2.5f * (yc - 1.2f);
                float bs = boxT[(wv + 8 * rs) * TILE_W + pm] * (1.0f / 25.0f);
                float inorm = 0.85f - 2.0f * bs * bs;
                float yn = yo + 0.01f * (-yo + gxo + accW[rs][v] + 1.0f);
                float xn = xo + 0.01f * ((0.8f * gxo + accJ[rs][v] + in_v + inorm)
                                         - (xo + gyo + accP[rs][v]));
                float gxn = fminf(fmaxf(xn - 1.0f, 0.0f), 1.0f);
                xs[ci] = xn;
                ys[ci] = yn;
                outAcc[ci] += gxn;
                gxOut[pidx * KP + n] = (_Float16)gxn;
            }
        }
    }
}

// ---------------------------------------------------------------------------
// out[b,h,w] = max_c out_acc / STEPS
// ---------------------------------------------------------------------------
__global__ __launch_bounds__(256) void final_kernel(
    const float* __restrict__ outAcc, float* __restrict__ out)
{
    size_t i = (size_t)blockIdx.x * 256 + threadIdx.x;
    if (i >= NPIX) return;
    float mx = outAcc[i * KC];
    for (int c = 1; c < KC; c++) mx = fmaxf(mx, outAcc[i * KC + c]);
    out[i] = mx * (1.0f / 16.0f);
}

extern "C" void kernel_launch(void* const* d_in, const int* in_sizes, int n_in,
                              void* d_out, int out_size, void* d_ws, size_t ws_size,
                              hipStream_t stream) {
    const float* inp  = (const float*)d_in[0];  // (2,512,512,12)
    const float* Wsrc = (const float*)d_in[1];  // (15,15,12,12)
    const float* Jsrc = (const float*)d_in[2];  // (15,15,12,12)
    const float* Psrc = (const float*)d_in[3];  // (1,1,12,12)
    float* out = (float*)d_out;

    char* w = (char*)d_ws;
    auto alloc = [&](size_t bytes) {
        char* p = w;
        w += (bytes + 255) & ~(size_t)255;
        return p;
    };
    float*     xs     = (float*)alloc(NPIX * KC * sizeof(float));
    float*     ys     = (float*)alloc(NPIX * KC * sizeof(float));
    float*     outAcc = (float*)alloc(NPIX * KC * sizeof(float));
    _Float16*  gxA    = (_Float16*)alloc(NPIX * KP * sizeof(_Float16));
    _Float16*  gxB    = (_Float16*)alloc(NPIX * KP * sizeof(_Float16));
    _Float16*  Wpk    = (_Float16*)alloc(WPK_ELEMS * sizeof(_Float16));
    _Float16*  Jpk    = (_Float16*)alloc(WPK_ELEMS * sizeof(_Float16));
    _Float16*  Ppk    = (_Float16*)alloc(512 * sizeof(_Float16));

    prep_kernel<<<(2 * WPK_ELEMS + 512 + 255) / 256, 256, 0, stream>>>(
        Wsrc, Jsrc, Psrc, Wpk, Jpk, Ppk);
    init_kernel<<<(int)((NPIX + 255) / 256), 256, 0, stream>>>(
        xs, ys, outAcc, gxA, gxB);

    dim3 grid(WW / TILE_W, HH / TILE_H, BATCH);
    for (int s = 0; s < 16; s++) {
        const _Float16* gin  = (s & 1) ? gxB : gxA;
        _Float16*       gout = (s & 1) ? gxA : gxB;
        step_kernel<<<grid, 256, 0, stream>>>(
            gin, gout, xs, ys, outAcc, inp, Wpk, Jpk, Ppk);
    }
    final_kernel<<<(int)((NPIX + 255) / 256), 256, 0, stream>>>(outAcc, out);
}
